// multiheadattention_14319420965213
// MI455X (gfx1250) — matrix operand order repaired and checked by reading, not compiled, not run on hardware
//
#include <hip/hip_runtime.h>

// ---------------------------------------------------------------------------
// Multi-head attention forward (B=2, S=2048, D=512, H=8, DK=64) for gfx1250.
// All matmuls: LDS-tiled GEMM, TDM (tensor_load_to_lds) double-buffered
// staging, v_wmma_f32_16x16x32_bf16 with 4 accumulator tiles per wave.
// Outputs: d_out = [ out fp32 (B*S*D) | p_attn fp32 (B*H*S*S) ].
// ---------------------------------------------------------------------------

typedef __attribute__((ext_vector_type(16))) __bf16 v16bf;
typedef __attribute__((ext_vector_type(8)))  __bf16 v8bf;
typedef __attribute__((ext_vector_type(8)))  float  v8f;
typedef __attribute__((ext_vector_type(4)))  unsigned int v4u;
typedef __attribute__((ext_vector_type(8)))  unsigned int v8u;

constexpr int B  = 2;
constexpr int S  = 2048;
constexpr int D  = 512;
constexpr int H  = 8;
constexpr int DK = 64;           // D / H
constexpr int BS = B * S;        // 4096

// Block tile geometry: 128 x 64 output per block, BK = 32 (one WMMA K-step).
// 8 waves: 2 (M) x 4 (N); each wave owns 4 stacked 16x16 accumulator tiles.
constexpr int BM = 128;
constexpr int BN = 64;
constexpr int BK = 32;

// Output modes for the tiled GEMM epilogue.
constexpr int OUT_F32  = 0;  // fp32 [M,N], scaled (scores, final proj)
constexpr int OUT_HEAD = 1;  // bf16 head-major [B,H,S,DK]      (Q,K proj)
constexpr int OUT_VT   = 2;  // bf16 head-transposed [B,H,DK,S] (V proj)
constexpr int OUT_CTX  = 3;  // bf16 merged-head [B,S,D]        (P @ V)

// ---------------------------------------------------------------------------
// fp32 -> bf16 elementwise conversion
// ---------------------------------------------------------------------------
__global__ __launch_bounds__(256) void cvt_f32_bf16(const float* __restrict__ x,
                                                    __bf16* __restrict__ y, int n) {
  int i = blockIdx.x * 256 + threadIdx.x;
  if (i < n) y[i] = (__bf16)x[i];
}

// ---------------------------------------------------------------------------
// Tensor Data Mover: 2D tile (tile_rows x tile_k elems) global -> LDS.
// Builds the D# per CDNA5 ISA (group0 128b + group1 256b) and issues the
// 2-group form of tensor_load_to_lds. All inputs are block-uniform -> SGPRs.
//   dsz_code: 0=1B, 1=2B, 2=4B elements
//   row_stride: elements between consecutive rows in global memory
// ---------------------------------------------------------------------------
__device__ __forceinline__ void tdm_load_tile(unsigned lds_off, const void* gptr,
                                              unsigned dsz_code,
                                              unsigned tile_k, unsigned tile_rows,
                                              unsigned long row_stride,
                                              unsigned long tdim0) {
  const unsigned long ga = (unsigned long)gptr;
  v4u g0;
  g0.x = 1u;                                        // [1:0] count=1 (valid)
  g0.y = lds_off;                                   // [63:32] lds_addr (bytes)
  g0.z = (unsigned)(ga & 0xFFFFFFFFu);              // global_addr[31:0]
  g0.w = (unsigned)((ga >> 32) & 0x01FFFFFFu)       // global_addr[56:32]
       | (2u << 30);                                // [127:126] type = 2
  v8u g1;
  g1[0] = dsz_code << 16;                           // [17:16] data_size
  g1[1] = (unsigned)((tdim0 & 0xFFFFu) << 16);      // tensor_dim0[15:0]
  g1[2] = (unsigned)((tdim0 >> 16) & 0xFFFFu)       // tensor_dim0[31:16]
        | ((tile_rows & 0xFFFFu) << 16);            // tensor_dim1[15:0]
  g1[3] = (tile_k << 16);                           // [127:112] tile_dim0
  g1[4] = tile_rows;                                // [143:128] tile_dim1
  g1[5] = (unsigned)(row_stride & 0xFFFFFFFFu);     // tensor_dim0_stride[31:0]
  g1[6] = (unsigned)((row_stride >> 32) & 0xFFFFu); // tensor_dim0_stride[47:32]
  g1[7] = 0u;
  asm volatile("tensor_load_to_lds %0, %1" : : "s"(g0), "s"(g1) : "memory");
}

// A-fragment from an LDS tile row (32 elements), per ISA 16-bit A layout:
// lanes 0-15 hold K=[ka..ka+7]+[16+ka..], lanes 16-31 ka=8.
__device__ __forceinline__ v16bf load_afrag(const __bf16* row, int ka) {
  v8bf a0 = *(const v8bf*)(row + ka);
  v8bf a1 = *(const v8bf*)(row + 16 + ka);
  return __builtin_shufflevector(a0, a1, 0, 1, 2, 3, 4, 5, 6, 7,
                                 8, 9, 10, 11, 12, 13, 14, 15);
}
__device__ __forceinline__ v16bf load_afrag(const float* row, int ka) {
  v16bf a;
#pragma unroll
  for (int i = 0; i < 8; ++i) {
    a[i]     = (__bf16)row[ka + i];       // fp32 LDS -> bf16 in-register
    a[i + 8] = (__bf16)row[16 + ka + i];
  }
  return a;
}

// ---------------------------------------------------------------------------
// Tiled, batched C = A * W^T (bf16 WMMA, fp32 accumulate), TDM-staged LDS,
// double buffered. A: [M,Kd] (AT = __bf16 or float), W: [N,Kd] bf16 row-major.
// ---------------------------------------------------------------------------
template <typename AT, int OUTMODE>
__global__ __launch_bounds__(256) void gemm_tiled(
    const AT* __restrict__ A, long sA,
    const __bf16* __restrict__ W, long sW,
    float* __restrict__ Cf, long sC,
    __bf16* __restrict__ Cb,
    int N, int Kd, int ntn, float scale) {
  __shared__ AT     As[2][BM * BK];
  __shared__ __bf16 Bs[2][BN * BK];

  const int z = blockIdx.y;
  const AT*     Ab = A + (long)z * sA;
  const __bf16* Wb = W + (long)z * sW;

  const int bm = blockIdx.x / ntn;
  const int bn = blockIdx.x % ntn;

  const int lane = threadIdx.x & 31;
  const int wave = threadIdx.x >> 5;
  const int wm   = wave >> 2;           // 0..1 (M direction, 64 rows each)
  const int wn   = wave & 3;            // 0..3 (N direction, 16 cols each)

  const int r  = lane & 15;
  const int ka = (lane & 16) ? 8 : 0;
  const int kb = (lane & 16) ? 16 : 0;

  const unsigned ldsA[2] = {(unsigned)(unsigned long)&As[0][0],
                            (unsigned)(unsigned long)&As[1][0]};
  const unsigned ldsB[2] = {(unsigned)(unsigned long)&Bs[0][0],
                            (unsigned)(unsigned long)&Bs[1][0]};
  constexpr unsigned dszA = (sizeof(AT) == 4) ? 2u : 1u;

  const AT*     Ag = Ab + (long)(bm * BM) * Kd;   // tile row block base
  const __bf16* Wg = Wb + (long)(bn * BN) * Kd;

  const int nk = Kd / BK;

  // Prologue: stage k-step 0 into buffer 0 (wave 0 drives the TDM).
  if (wave == 0) {
    tdm_load_tile(ldsA[0], Ag, dszA, BK, BM, (unsigned long)Kd, (unsigned long)Kd);
    tdm_load_tile(ldsB[0], Wg, 1u,  BK, BN, (unsigned long)Kd, (unsigned long)Kd);
  }

  v8f acc[4] = {v8f{}, v8f{}, v8f{}, v8f{}};

  for (int ks = 0; ks < nk; ++ks) {
    const int buf = ks & 1;
    if (wave == 0) {
      if (ks + 1 < nk) {
        // Issue next stage into the other buffer, then wait for the two
        // current-stage loads (tensor ops complete in order per wave).
        const long koff = (long)(ks + 1) * BK;
        tdm_load_tile(ldsA[buf ^ 1], Ag + koff, dszA, BK, BM,
                      (unsigned long)Kd, (unsigned long)Kd);
        tdm_load_tile(ldsB[buf ^ 1], Wg + koff, 1u, BK, BN,
                      (unsigned long)Kd, (unsigned long)Kd);
        __builtin_amdgcn_s_wait_tensorcnt(2);
      } else {
        __builtin_amdgcn_s_wait_tensorcnt(0);
      }
    }
    __syncthreads();   // staged tile visible to all 8 waves

    // B fragment: column (bn*64 + wn*16 + r) of W^T = LDS row, K=[ka..ka+7]+[16+ka..].
    const __bf16* Brow = &Bs[buf][(wn * 16 + r) * BK];
    v8bf b0 = *(const v8bf*)(Brow + ka);
    v8bf b1 = *(const v8bf*)(Brow + 16 + ka);
    v16bf bfrag = __builtin_shufflevector(b0, b1, 0, 1, 2, 3, 4, 5, 6, 7,
                                          8, 9, 10, 11, 12, 13, 14, 15);
#pragma unroll
    for (int t = 0; t < 4; ++t) {
      const AT* Arow = &As[buf][(wm * 64 + t * 16 + r) * BK];
      v16bf afrag = load_afrag(Arow, ka);
      acc[t] = __builtin_amdgcn_wmma_f32_16x16x32_bf16(
          false, afrag, false, bfrag, (short)0, acc[t], false, false);
    }
    __syncthreads();   // done reading before this buffer is overwritten
  }

  // Epilogue: C/D layout -> lane n<16: col n rows m0..m0+7; lane>=16: +8 rows.
  const int n = bn * BN + wn * 16 + r;
#pragma unroll
  for (int t = 0; t < 4; ++t) {
    const int m0 = bm * BM + wm * 64 + t * 16 + ((lane & 16) ? 8 : 0);
    if constexpr (OUTMODE == OUT_F32) {
      float* Cz = Cf + (long)z * sC;
#pragma unroll
      for (int i = 0; i < 8; ++i)
        Cz[(long)(m0 + i) * N + n] = acc[t][i] * scale;
    } else if constexpr (OUTMODE == OUT_HEAD) {
      const int h = n >> 6, dk = n & (DK - 1);
#pragma unroll
      for (int i = 0; i < 8; ++i) {
        const int m = m0 + i, bb = m >> 11, s = m & (S - 1);
        Cb[(((long)bb * H + h) * S + s) * DK + dk] = (__bf16)acc[t][i];
      }
    } else if constexpr (OUTMODE == OUT_VT) {
      const int h = n >> 6, dk = n & (DK - 1);
#pragma unroll
      for (int i = 0; i < 8; ++i) {
        const int m = m0 + i, bb = m >> 11, s = m & (S - 1);
        Cb[(((long)bb * H + h) * DK + dk) * S + s] = (__bf16)acc[t][i];
      }
    } else {  // OUT_CTX: z = b*H+h, N == DK, n is dk
      const int bb = z >> 3, h = z & (H - 1);
#pragma unroll
      for (int i = 0; i < 8; ++i)
        Cb[((long)bb * S + (m0 + i)) * D + h * DK + n] = (__bf16)acc[t][i];
    }
  }
}

// ---------------------------------------------------------------------------
// Row softmax in place: one 256-thread block per row of `ncols` fp32.
// ---------------------------------------------------------------------------
__global__ __launch_bounds__(256) void softmax_rows(float* __restrict__ p, int ncols) {
  float* row = p + (long)blockIdx.x * ncols;
  __shared__ float red[256];
  const int t = threadIdx.x;

  float m = -__builtin_huge_valf();
  for (int i = t; i < ncols; i += 256) m = fmaxf(m, row[i]);
  red[t] = m;
  __syncthreads();
  for (int s2 = 128; s2 > 0; s2 >>= 1) {
    if (t < s2) red[t] = fmaxf(red[t], red[t + s2]);
    __syncthreads();
  }
  m = red[0];
  __syncthreads();

  float sum = 0.f;
  for (int i = t; i < ncols; i += 256) {
    float e = __expf(row[i] - m);
    row[i] = e;
    sum += e;
  }
  red[t] = sum;
  __syncthreads();
  for (int s2 = 128; s2 > 0; s2 >>= 1) {
    if (t < s2) red[t] += red[t + s2];
    __syncthreads();
  }
  const float inv = 1.0f / red[0];
  for (int i = t; i < ncols; i += 256) row[i] *= inv;
}

// ---------------------------------------------------------------------------
extern "C" void kernel_launch(void* const* d_in, const int* in_sizes, int n_in,
                              void* d_out, int out_size, void* d_ws, size_t ws_size,
                              hipStream_t stream) {
  const float* q_in = (const float*)d_in[0];
  const float* k_in = (const float*)d_in[1];
  const float* v_in = (const float*)d_in[2];
  const float* w_q  = (const float*)d_in[3];
  const float* w_k  = (const float*)d_in[4];
  const float* w_v  = (const float*)d_in[5];
  const float* w_o  = (const float*)d_in[6];

  float* out   = (float*)d_out;              // [B, S, D]
  float* pattn = out + (long)BS * D;         // [B, H, S, S]

  const long nQKV = (long)BS * D;            // 2,097,152
  const long nW   = (long)D * D;             //   262,144
  char* ws = (char*)d_ws;
  __bf16* qb  = (__bf16*)ws;  ws += nQKV * 2;
  __bf16* kb  = (__bf16*)ws;  ws += nQKV * 2;
  __bf16* vb  = (__bf16*)ws;  ws += nQKV * 2;
  __bf16* wqb = (__bf16*)ws;  ws += nW * 2;
  __bf16* wkb = (__bf16*)ws;  ws += nW * 2;
  __bf16* wvb = (__bf16*)ws;  ws += nW * 2;
  __bf16* wob = (__bf16*)ws;  ws += nW * 2;
  __bf16* Qh  = (__bf16*)ws;  ws += nQKV * 2;   // [B,H,S,DK]
  __bf16* Kh  = (__bf16*)ws;  ws += nQKV * 2;   // [B,H,S,DK]
  __bf16* Vt  = (__bf16*)ws;  ws += nQKV * 2;   // [B,H,DK,S]  (transposed V)
  __bf16* ctx = (__bf16*)ws;  ws += nQKV * 2;   // [B,S,D] merged heads

  // 1) fp32 -> bf16 conversions
  cvt_f32_bf16<<<(int)(nQKV / 256), 256, 0, stream>>>(q_in, qb, (int)nQKV);
  cvt_f32_bf16<<<(int)(nQKV / 256), 256, 0, stream>>>(k_in, kb, (int)nQKV);
  cvt_f32_bf16<<<(int)(nQKV / 256), 256, 0, stream>>>(v_in, vb, (int)nQKV);
  cvt_f32_bf16<<<(int)(nW / 256), 256, 0, stream>>>(w_q, wqb, (int)nW);
  cvt_f32_bf16<<<(int)(nW / 256), 256, 0, stream>>>(w_k, wkb, (int)nW);
  cvt_f32_bf16<<<(int)(nW / 256), 256, 0, stream>>>(w_v, wvb, (int)nW);
  cvt_f32_bf16<<<(int)(nW / 256), 256, 0, stream>>>(w_o, wob, (int)nW);

  // 2) Projections: [4096,512] @ W^T. (4096/128)*(512/64) = 256 blocks.
  gemm_tiled<__bf16, OUT_HEAD><<<dim3(256, 1), 256, 0, stream>>>(
      qb, 0, wqb, 0, nullptr, 0, Qh, D, D, /*ntn=*/8, 1.0f);
  gemm_tiled<__bf16, OUT_HEAD><<<dim3(256, 1), 256, 0, stream>>>(
      kb, 0, wkb, 0, nullptr, 0, Kh, D, D, 8, 1.0f);
  gemm_tiled<__bf16, OUT_VT><<<dim3(256, 1), 256, 0, stream>>>(
      vb, 0, wvb, 0, nullptr, 0, Vt, D, D, 8, 1.0f);

  // 3) Scores = (Q K^T)/sqrt(DK), batched over B*H.
  //    (2048/128)*(2048/64) = 512 blocks per head.
  gemm_tiled<__bf16, OUT_F32><<<dim3(512, B * H), 256, 0, stream>>>(
      Qh, (long)S * DK, Kh, (long)S * DK, pattn, (long)S * S, nullptr,
      S, DK, /*ntn=*/32, 0.125f);

  // 4) Row softmax in place over the 2048-wide rows of p_attn.
  softmax_rows<<<B * H * S, 256, 0, stream>>>(pattn, S);

  // 5) ctx = P @ V = P @ Vt^T, batched over heads; fp32 A converted in LDS.
  //    (2048/128)*(64/64) = 16 blocks per head.
  gemm_tiled<float, OUT_CTX><<<dim3(16, B * H), 256, 0, stream>>>(
      pattn, (long)S * S, Vt, (long)DK * S, nullptr, 0, ctx,
      DK, S, /*ntn=*/1, 1.0f);

  // 6) out = ctx @ w_o^T, fp32 into d_out.
  gemm_tiled<__bf16, OUT_F32><<<dim3(256, 1), 256, 0, stream>>>(
      ctx, 0, wob, 0, out, 0, nullptr, D, D, 8, 1.0f);
}
